// GCN_ConvE_24635932410316
// MI455X (gfx1250) — compile-verified, matrix operand order checked by reading, not run on hardware
//
#include <hip/hip_runtime.h>
#include <hip/hip_bf16.h>
#include <math.h>

// ---------------------------------------------------------------------------
// CompGCN(2 layers) + ConvE head for MI455X (gfx1250, wave32, WMMA).
// Heavy GEMMs use v_wmma_f32_16x16x32_f16 (f16 in / f32 accumulate).
// Edge message matmul is algebraically hoisted past the scatter-sum:
//   segment_sum(enorm*(x[src]*r[et]) @ W) == segment_sum(enorm*comp) @ W
// cutting GEMM work 5x; scatter atomics hit L2 (80MB accumulator << 192MB L2).
// FC GEMM stages its B tile with GLOBAL_LOAD_ASYNC_TO_LDS_B128 (ASYNCcnt).
// ---------------------------------------------------------------------------

typedef _Float16 v16h __attribute__((ext_vector_type(16)));
typedef float    v8f  __attribute__((ext_vector_type(8)));
typedef int      vi4  __attribute__((vector_size(4 * sizeof(int))));

#define NUM_ENT 100000
#define DIM     200
#define NE      1000000
#define NB      1024
#define KCAT    608      // 600 (=3*DIM concat) padded to multiple of 32
#define NFILT   200
#define FLATSZ  39200    // 200*14*14, already a multiple of 32
#define FC_KCH  1024     // K-chunk staged in LDS for the FC GEMM

#define AS1 __attribute__((address_space(1)))
#define AS3 __attribute__((address_space(3)))

#if defined(__has_builtin)
#  if __has_builtin(__builtin_amdgcn_global_load_async_to_lds_b128) && \
      __has_builtin(__builtin_amdgcn_s_wait_asynccnt)
#    define USE_ASYNC_LDS 1
#  endif
#endif
#ifndef USE_ASYNC_LDS
#  define USE_ASYNC_LDS 0
#endif

static __device__ __forceinline__ void atomAdd(float* p, float v) {
  __hip_atomic_fetch_add(p, v, __ATOMIC_RELAXED, __HIP_MEMORY_SCOPE_AGENT);
}

// ---------------------------------------------------------------------------
// 1) Edge scatter: P[dst] += enorm * x[src] * r[etype]   (split in/out halves)
//    One wave per edge; x and r rows are L2 resident.
// ---------------------------------------------------------------------------
__global__ void __launch_bounds__(256) edge_scatter(
    const float* __restrict__ x, const float* __restrict__ r,
    const int* __restrict__ src, const int* __restrict__ dst,
    const int* __restrict__ etype, const float* __restrict__ enorm,
    float* __restrict__ Pin, float* __restrict__ Pout) {
  int e = blockIdx.x * 8 + (threadIdx.x >> 5);
  if (e >= NE) return;
  int lane = threadIdx.x & 31;
  int s = src[e], d = dst[e], t = etype[e];
  float nrm = enorm[e];
  float* P = (e < NE / 2) ? Pin : Pout;
  const float* xr = x + (size_t)s * DIM;
  const float* rr = r + (size_t)t * DIM;
  float* pr = P + (size_t)d * DIM;
  for (int c = lane; c < DIM; c += 32)
    atomAdd(&pr[c], xr[c] * rr[c] * nrm);
}

// ---------------------------------------------------------------------------
// 2) Pack A = [Pin | Pout | x*loop_rel] (K padded 600->608) as f16 rows.
// ---------------------------------------------------------------------------
__global__ void __launch_bounds__(256) pack_a(
    const float* __restrict__ Pin, const float* __restrict__ Pout,
    const float* __restrict__ x, const float* __restrict__ loop_rel,
    _Float16* __restrict__ Apack) {
  int row = blockIdx.x;
  for (int j = threadIdx.x; j < KCAT; j += 256) {
    float v = 0.f;
    if (j < 200)      v = Pin[(size_t)row * DIM + j];
    else if (j < 400) v = Pout[(size_t)row * DIM + (j - 200)];
    else if (j < 600) v = x[(size_t)row * DIM + (j - 400)] * loop_rel[j - 400];
    Apack[(size_t)row * KCAT + j] = (_Float16)v;
  }
}

// ---------------------------------------------------------------------------
// 3) GCN GEMM: H = (A[100000x608] @ Wcat[608x200]) / 3 + bias
//    Wcat = [w_in; w_out; w_loop] stacked on K. One 16-col N-tile per block,
//    B^T staged in LDS as f16; 8 waves x 16 M-rows each; 19 WMMA k-steps.
//    Row index is CLAMPED (not predicated) so the unrolled main loop has no
//    exec-mask manipulation and no fragment re-zeroing; OOB M rows are simply
//    not stored.
// ---------------------------------------------------------------------------
__global__ void __launch_bounds__(256) gcn_gemm(
    const _Float16* __restrict__ Apack,
    const float* __restrict__ w_in, const float* __restrict__ w_out,
    const float* __restrict__ w_loop, const float* __restrict__ bias,
    float* __restrict__ H) {
  __shared__ __align__(32) _Float16 Bt[16][KCAT];   // [n][k] = Wcat[k][col]
  const int nt = blockIdx.x;                        // 0..12
  const int mbase = blockIdx.y * 128;
  // Stage B^T tile (f32 -> f16); nested loops avoid integer div/mod.
  for (int n = 0; n < 16; ++n) {
    int c = nt * 16 + n;
    for (int k = threadIdx.x; k < KCAT; k += 256) {
      float v = 0.f;
      if (c < DIM && k < 600) {
        if (k < 200)      v = w_in[k * DIM + c];
        else if (k < 400) v = w_out[(k - 200) * DIM + c];
        else              v = w_loop[(k - 400) * DIM + c];
      }
      Bt[n][k] = (_Float16)v;
    }
  }
  __syncthreads();
  const int lane = threadIdx.x & 31;
  const int wave = threadIdx.x >> 5;
  const int rowA = mbase + wave * 16 + (lane & 15);
  const int rowL = min(rowA, NUM_ENT - 1);          // clamp: always a valid row
  const int khalf = (lane >> 4) << 4;               // lanes 16-31 hold K+16..31
  const _Float16* ap = Apack + (size_t)rowL * KCAT + khalf;
  const _Float16* bp = &Bt[lane & 15][khalf];
  v8f acc = {};
#pragma unroll
  for (int kt = 0; kt < KCAT / 32; ++kt) {
    v16h a = *(const v16h*)(ap + kt * 32);
    v16h b = *(const v16h*)(bp + kt * 32);
    acc = __builtin_amdgcn_wmma_f32_16x16x32_f16(false, a, false, b,
                                                 (short)0, acc, false, false);
  }
  const int colC = nt * 16 + (lane & 15);
  const int rbase = mbase + wave * 16 + ((lane >> 4) << 3);
  if (colC < DIM) {
    float bv = bias[colC];
#pragma unroll
    for (int i = 0; i < 8; ++i) {
      int rr = rbase + i;
      if (rr < NUM_ENT) H[(size_t)rr * DIM + colC] = acc[i] * (1.f / 3.f) + bv;
    }
  }
}

// ---------------------------------------------------------------------------
// Column-wise sum/sumsq reduction (BatchNorm1d statistics).
// ---------------------------------------------------------------------------
__global__ void __launch_bounds__(256) col_stats(
    const float* __restrict__ A, int rows, int cols,
    float* __restrict__ sum, float* __restrict__ sumsq) {
  int r0 = blockIdx.x * 64;
  int rend = min(r0 + 64, rows);
  for (int c = threadIdx.x; c < cols; c += blockDim.x) {
    float s = 0.f, q = 0.f;
    for (int r = r0; r < rend; ++r) {
      float v = A[(size_t)r * cols + c];
      s += v; q += v * v;
    }
    atomAdd(&sum[c], s);
    atomAdd(&sumsq[c], q);
  }
}

__global__ void __launch_bounds__(256) bn_tanh(
    float* __restrict__ H, const float* __restrict__ sum,
    const float* __restrict__ sumsq, const float* __restrict__ g,
    const float* __restrict__ be, int rows, int cols) {
  int idx = blockIdx.x * 256 + threadIdx.x;
  if (idx >= rows * cols) return;
  int c = idx % cols;
  float invN = 1.f / (float)rows;
  float m = sum[c] * invN;
  float var = sumsq[c] * invN - m * m;
  float v = (H[idx] - m) * rsqrtf(var + 1e-5f) * g[c] + be[c];
  H[idx] = tanhf(v);
}

// r_out = r_in @ w_rel  (200x200 @ 200x200 — negligible, naive)
__global__ void __launch_bounds__(256) rel_mm(
    const float* __restrict__ Rin, const float* __restrict__ W,
    float* __restrict__ Rout) {
  int idx = blockIdx.x * 256 + threadIdx.x;
  if (idx >= 200 * DIM) return;
  int i = idx / DIM, c = idx % DIM;
  float acc = 0.f;
  for (int k = 0; k < DIM; ++k) acc += Rin[i * DIM + k] * W[k * DIM + c];
  Rout[idx] = acc;
}

// ---------------------------------------------------------------------------
// ConvE head
// ---------------------------------------------------------------------------
__global__ void __launch_bounds__(256) gather_head(
    const float* __restrict__ x, const float* __restrict__ r,
    const int* __restrict__ subj, const int* __restrict__ relIdx,
    const int* __restrict__ obj, float* __restrict__ stk,
    float* __restrict__ objT) {
  int idx = blockIdx.x * 256 + threadIdx.x;
  if (idx >= NB * DIM) return;
  int b = idx / DIM, c = idx % DIM;
  stk[b * 400 + c]       = x[(size_t)subj[b] * DIM + c];
  stk[b * 400 + 200 + c] = r[(size_t)relIdx[b] * DIM + c];
  objT[c * NB + b]       = x[(size_t)obj[b] * DIM + c];   // transposed for scoring
}

__global__ void bn0_finalize(float* stats) {              // C=1 -> reduce 400 cols
  __shared__ float rs[512], rq[512];
  int t = threadIdx.x;
  rs[t] = (t < 400) ? stats[t] : 0.f;
  rq[t] = (t < 400) ? stats[512 + t] : 0.f;
  __syncthreads();
  for (int o = 256; o > 0; o >>= 1) {
    if (t < o) { rs[t] += rs[t + o]; rq[t] += rq[t + o]; }
    __syncthreads();
  }
  if (t == 0) { stats[1024] = rs[0]; stats[1025] = rq[0]; }
}

__global__ void __launch_bounds__(256) bn0_norm(
    float* __restrict__ stk, const float* __restrict__ stats,
    const float* __restrict__ g, const float* __restrict__ b) {
  int idx = blockIdx.x * 256 + threadIdx.x;
  if (idx >= NB * 400) return;
  const float invN = 1.f / (NB * 400.f);
  float m = stats[1024] * invN;
  float var = stats[1025] * invN - m * m;
  stk[idx] = (stk[idx] - m) * rsqrtf(var + 1e-5f) * g[0] + b[0];
}

// 7x7 VALID conv, 200 filters, 20x20 input -> 14x14. Image + all filters in LDS.
__global__ void __launch_bounds__(256) conv7(
    const float* __restrict__ stk, const float* __restrict__ cw,
    const float* __restrict__ cb, float* __restrict__ out) {
  __shared__ float simg[400];
  __shared__ float sw[NFILT * 49];
  int b = blockIdx.y;
  for (int i = threadIdx.x; i < 400; i += 256) simg[i] = stk[b * 400 + i];
  for (int i = threadIdx.x; i < NFILT * 49; i += 256) sw[i] = cw[i];
  __syncthreads();
  int p = blockIdx.x * 256 + threadIdx.x;                 // over 200*196
  if (p >= NFILT * 196) return;
  int f = p / 196, q = p % 196, y = q / 14, xx = q % 14;
  float acc = cb[f];
  const float* w = sw + f * 49;
#pragma unroll
  for (int ky = 0; ky < 7; ++ky)
#pragma unroll
    for (int kx = 0; kx < 7; ++kx)
      acc += simg[(y + ky) * 20 + (xx + kx)] * w[ky * 7 + kx];
  out[(size_t)b * FLATSZ + p] = acc;
}

// Per-channel stats over (B, 14, 14): 98 blocks * 2048 elems = 200704 exactly.
__global__ void __launch_bounds__(256) bn1_stats(
    const float* __restrict__ conv, float* __restrict__ sum,
    float* __restrict__ sumsq) {
  int f = blockIdx.x;
  float s = 0.f, q = 0.f;
  for (int k = 0; k < 8; ++k) {
    int e = blockIdx.y * 2048 + k * 256 + threadIdx.x;
    int b = e / 196, p = e % 196;
    float v = conv[(size_t)b * FLATSZ + f * 196 + p];
    s += v; q += v * v;
  }
  __shared__ float rs[256], rq[256];
  rs[threadIdx.x] = s; rq[threadIdx.x] = q;
  __syncthreads();
  for (int o = 128; o > 0; o >>= 1) {
    if (threadIdx.x < o) { rs[threadIdx.x] += rs[threadIdx.x + o]; rq[threadIdx.x] += rq[threadIdx.x + o]; }
    __syncthreads();
  }
  if (threadIdx.x == 0) { atomAdd(&sum[f], rs[0]); atomAdd(&sumsq[f], rq[0]); }
}

// BN1 + ReLU, emit f16 so the FC GEMM A-operand is a straight b128 load.
__global__ void __launch_bounds__(256) bn1_relu_cvt(
    const float* __restrict__ conv, const float* __restrict__ sum,
    const float* __restrict__ sumsq, const float* __restrict__ g,
    const float* __restrict__ be, _Float16* __restrict__ out16) {
  int idx = blockIdx.x * 256 + threadIdx.x;
  if (idx >= NB * FLATSZ) return;
  int f = (idx / 196) % NFILT;
  const float invN = 1.f / (NB * 196.f);
  float m = sum[f] * invN;
  float var = sumsq[f] * invN - m * m;
  float h = (conv[idx] - m) * rsqrtf(var + 1e-5f) * g[f] + be[f];
  out16[idx] = (_Float16)fmaxf(h, 0.f);
}

__global__ void __launch_bounds__(256) fcw_cvt(
    const float* __restrict__ w, _Float16* __restrict__ w16) {
  int idx = blockIdx.x * 256 + threadIdx.x;
  if (idx < DIM * FLATSZ) w16[idx] = (_Float16)w[idx];
}

// ---------------------------------------------------------------------------
// FC GEMM: [1024 x 39200] @ fc_w^T -> [1024 x 200]; fc_w is [200][39200] so it
// is already in B^T ([n][k]) layout. K streamed in 1024-wide LDS chunks.
// B staging uses GLOBAL_LOAD_ASYNC_TO_LDS_B128 (waited via s_wait_asynccnt);
// the builtin takes pointers to 16-byte int vectors in AS1/AS3. Padded
// columns >= 200 read in-workspace garbage that is masked at the store.
// ---------------------------------------------------------------------------
__global__ void __launch_bounds__(256) fc_gemm(
    const _Float16* __restrict__ A, const _Float16* __restrict__ BT,
    const float* __restrict__ fc_b, float* __restrict__ Cout) {
  __shared__ __align__(32) _Float16 Bs[16][FC_KCH];
  const int nt = blockIdx.x;                       // 0..12
  const int mbase = blockIdx.y * 128;              // 1024/128 = 8 blocks exact
  const int lane = threadIdx.x & 31;
  const int wave = threadIdx.x >> 5;
  const int rowA = mbase + wave * 16 + (lane & 15);
  const int khalf = (lane >> 4) << 4;
  const _Float16* ap = A + (size_t)rowA * FLATSZ + khalf;
  v8f acc = {};
  for (int kc = 0; kc < FLATSZ; kc += FC_KCH) {
    int kl = min(FC_KCH, FLATSZ - kc);             // always multiple of 32
#if USE_ASYNC_LDS
    for (int idx = threadIdx.x; idx < 2048; idx += 256) {   // 2048 b128 copies
      int n = idx >> 7;                            // 128 transfers per column
      int k8 = (idx & 127) << 3;                   // half-element offset
      const _Float16* gsrc = BT + (size_t)(nt * 16 + n) * FLATSZ + kc + k8;
      __builtin_amdgcn_global_load_async_to_lds_b128(
          (AS1 vi4*)gsrc, (AS3 vi4*)&Bs[n][k8], 0, 0);
    }
    __builtin_amdgcn_s_wait_asynccnt(0);
#else
    for (int idx = threadIdx.x; idx < 16 * FC_KCH; idx += 256) {
      int n = idx >> 10, k = idx & (FC_KCH - 1);
      int c = nt * 16 + n, kg = kc + k;
      _Float16 v = (_Float16)0.f;
      if (c < DIM && kg < FLATSZ) v = BT[(size_t)c * FLATSZ + kg];
      Bs[n][k] = v;
    }
#endif
    __syncthreads();
    if (kc + FC_KCH < FLATSZ)                      // prefetch next A chunk
      __builtin_prefetch(ap + kc + FC_KCH, 0, 1);
    int nks = kl / 32;
    for (int kt = 0; kt < nks; ++kt) {
      int k0 = kt * 32 + khalf;
      v16h a = *(const v16h*)(ap + kc + kt * 32);
      v16h b = *(const v16h*)(&Bs[lane & 15][k0]);
      acc = __builtin_amdgcn_wmma_f32_16x16x32_f16(false, a, false, b,
                                                   (short)0, acc, false, false);
    }
    __syncthreads();
  }
  const int colC = nt * 16 + (lane & 15);
  const int rbase = mbase + wave * 16 + ((lane >> 4) << 3);
  if (colC < DIM) {
    float bv = fc_b[colC];
#pragma unroll
    for (int i = 0; i < 8; ++i)
      Cout[(size_t)(rbase + i) * DIM + colC] = acc[i] + bv;
  }
}

__global__ void __launch_bounds__(256) bn2_relu(
    float* __restrict__ H, const float* __restrict__ sum,
    const float* __restrict__ sumsq, const float* __restrict__ g,
    const float* __restrict__ be) {
  int idx = blockIdx.x * 256 + threadIdx.x;
  if (idx >= NB * DIM) return;
  int c = idx % DIM;
  const float invN = 1.f / (float)NB;
  float m = sum[c] * invN;
  float var = sumsq[c] * invN - m * m;
  float v = (H[idx] - m) * rsqrtf(var + 1e-5f) * g[c] + be[c];
  H[idx] = fmaxf(v, 0.f);
}

// out[i][j] = sigmoid(dot(h2[i], obj[j])); objT transposed -> coalesced.
__global__ void __launch_bounds__(256) score(
    const float* __restrict__ h2, const float* __restrict__ objT,
    float* __restrict__ out) {
  int idx = blockIdx.x * 256 + threadIdx.x;
  if (idx >= NB * NB) return;
  int i = idx >> 10, j = idx & (NB - 1);
  float acc = 0.f;
  for (int c = 0; c < DIM; ++c) acc += h2[i * DIM + c] * objT[c * NB + j];
  out[idx] = 1.f / (1.f + __expf(-acc));
}

// ---------------------------------------------------------------------------
// Host orchestration (graph-capture safe: kernels + hipMemsetAsync only).
// ---------------------------------------------------------------------------
extern "C" void kernel_launch(void* const* d_in, const int* in_sizes, int n_in,
                              void* d_out, int out_size, void* d_ws, size_t ws_size,
                              hipStream_t stream) {
  (void)in_sizes; (void)n_in; (void)out_size; (void)ws_size;
  const float* init_embed = (const float*)d_in[0];
  const float* init_rel   = (const float*)d_in[1];
  const float* edge_norm  = (const float*)d_in[2];
  const float* w_in1   = (const float*)d_in[3];
  const float* w_out1  = (const float*)d_in[4];
  const float* w_loop1 = (const float*)d_in[5];
  const float* w_rel1  = (const float*)d_in[6];
  const float* lrel1   = (const float*)d_in[7];
  const float* b1  = (const float*)d_in[8];
  const float* g1  = (const float*)d_in[9];
  const float* be1 = (const float*)d_in[10];
  const float* w_in2   = (const float*)d_in[11];
  const float* w_out2  = (const float*)d_in[12];
  const float* w_loop2 = (const float*)d_in[13];
  const float* w_rel2  = (const float*)d_in[14];
  const float* lrel2   = (const float*)d_in[15];
  const float* b2  = (const float*)d_in[16];
  const float* g2  = (const float*)d_in[17];
  const float* be2 = (const float*)d_in[18];
  const float* conv_w = (const float*)d_in[19];
  const float* conv_b = (const float*)d_in[20];
  const float* fc_w   = (const float*)d_in[21];
  const float* fc_b   = (const float*)d_in[22];
  const float* bn0_g = (const float*)d_in[23];
  const float* bn0_b = (const float*)d_in[24];
  const float* bn1_g = (const float*)d_in[25];
  const float* bn1_b = (const float*)d_in[26];
  const float* bn2_g = (const float*)d_in[27];
  const float* bn2_b = (const float*)d_in[28];
  const int* subj    = (const int*)d_in[29];
  const int* rel_idx = (const int*)d_in[30];
  const int* obj     = (const int*)d_in[31];
  const int* src     = (const int*)d_in[32];
  const int* dst     = (const int*)d_in[33];
  const int* etype   = (const int*)d_in[34];

  // Workspace layout (bytes, all 32B-aligned); regions reused across phases.
  char* ws = (char*)d_ws;
  float*     X0     = (float*)(ws + 0ull);            // 80 MB  layer1 out
  float*     X1     = (float*)(ws + 80000000ull);     // 80 MB  layer2 out
  float*     PIN    = (float*)(ws + 160000000ull);    // 80 MB
  float*     POUT   = (float*)(ws + 240000000ull);    // 80 MB
  _Float16*  CONV16 = (_Float16*)(ws + 160000000ull); // reuses PIN/POUT in head
  _Float16*  APACK  = (_Float16*)(ws + 320000000ull); // 121.6 MB
  float*     CONVF  = (float*)(ws + 320000000ull);    // reuses APACK in head
  _Float16*  FCW16  = (_Float16*)(ws + 480640000ull); // 15.7 MB
  float*     R0     = (float*)(ws + 496320000ull);
  float*     R1     = (float*)(ws + 496640000ull);
  float*     STATS  = (float*)(ws + 496960000ull);    // sum[512]|sumsq[512]|totals
  float*     STK    = (float*)(ws + 496968192ull);    // 1024x400
  float*     OBJT   = (float*)(ws + 498606592ull);    // 200x1024
  float*     FCOUT  = (float*)(ws + 499425792ull);    // 1024x200

  auto layer = [&](const float* xin, const float* rin,
                   const float* wi, const float* wo, const float* wl,
                   const float* wr, const float* lrel, const float* bias,
                   const float* gg, const float* bb, float* xout, float* rout) {
    (void)hipMemsetAsync(PIN, 0, 160000000ull, stream);  // zero PIN+POUT (contiguous)
    (void)hipMemsetAsync(STATS, 0, 8192, stream);
    edge_scatter<<<NE / 8, 256, 0, stream>>>(xin, rin, src, dst, etype, edge_norm, PIN, POUT);
    pack_a<<<NUM_ENT, 256, 0, stream>>>(PIN, POUT, xin, lrel, APACK);
    gcn_gemm<<<dim3(13, (NUM_ENT + 127) / 128), 256, 0, stream>>>(APACK, wi, wo, wl, bias, xout);
    col_stats<<<(NUM_ENT + 63) / 64, 256, 0, stream>>>(xout, NUM_ENT, DIM, STATS, STATS + 512);
    bn_tanh<<<(NUM_ENT * DIM + 255) / 256, 256, 0, stream>>>(xout, STATS, STATS + 512, gg, bb, NUM_ENT, DIM);
    rel_mm<<<(200 * DIM + 255) / 256, 256, 0, stream>>>(rin, wr, rout);
  };

  layer(init_embed, init_rel, w_in1, w_out1, w_loop1, w_rel1, lrel1, b1, g1, be1, X0, R0);
  layer(X0,         R0,       w_in2, w_out2, w_loop2, w_rel2, lrel2, b2, g2, be2, X1, R1);

  // ---- ConvE head ----
  gather_head<<<(NB * DIM + 255) / 256, 256, 0, stream>>>(X1, R1, subj, rel_idx, obj, STK, OBJT);
  (void)hipMemsetAsync(STATS, 0, 8192, stream);
  col_stats<<<(NB + 63) / 64, 256, 0, stream>>>(STK, NB, 400, STATS, STATS + 512);
  bn0_finalize<<<1, 512, 0, stream>>>(STATS);
  bn0_norm<<<(NB * 400 + 255) / 256, 256, 0, stream>>>(STK, STATS, bn0_g, bn0_b);
  conv7<<<dim3((NFILT * 196 + 255) / 256, NB), 256, 0, stream>>>(STK, conv_w, conv_b, CONVF);
  (void)hipMemsetAsync(STATS, 0, 8192, stream);
  bn1_stats<<<dim3(NFILT, 98), 256, 0, stream>>>(CONVF, STATS, STATS + 512);
  bn1_relu_cvt<<<(NB * FLATSZ + 255) / 256, 256, 0, stream>>>(CONVF, STATS, STATS + 512, bn1_g, bn1_b, CONV16);
  fcw_cvt<<<(DIM * FLATSZ + 255) / 256, 256, 0, stream>>>(fc_w, FCW16);
  fc_gemm<<<dim3(13, NB / 128), 256, 0, stream>>>(CONV16, FCW16, fc_b, FCOUT);
  (void)hipMemsetAsync(STATS, 0, 8192, stream);
  col_stats<<<(NB + 63) / 64, 256, 0, stream>>>(FCOUT, NB, DIM, STATS, STATS + 512);
  bn2_relu<<<(NB * DIM + 255) / 256, 256, 0, stream>>>(FCOUT, STATS, STATS + 512, bn2_g, bn2_b);
  score<<<(NB * NB + 255) / 256, 256, 0, stream>>>(FCOUT, OBJT, (float*)d_out);
}
